// RGLRU_88021059764273
// MI455X (gfx1250) — matrix operand order repaired and checked by reading, not compiled, not run on hardware
//
#include <hip/hip_runtime.h>
#include <math.h>

typedef __bf16 bf16;
typedef bf16  v16bf __attribute__((ext_vector_type(16)));
typedef float v8f   __attribute__((ext_vector_type(8)));

#define T_LEN   8192
#define DM      768
#define DR      1024
#define CHUNK   256
#define NCHUNK  (T_LEN / CHUNK)

union FragAB { v16bf v; uint4 q[2]; };
union FragC  { v8f   v; float f[8]; };

__device__ __forceinline__ float gelu_tanh(float x) {
  return 0.5f * x * (1.0f + tanhf(0.7978845608028654f * (x + 0.044715f * x * x * x)));
}
__device__ __forceinline__ float sigm(float x) { return 1.0f / (1.0f + expf(-x)); }

// CDNA5 async global->LDS copy (ASYNCcnt-tracked, bypasses VGPRs).
__device__ __forceinline__ void async_b128(unsigned lds_off, const void* gptr) {
  asm volatile("global_load_async_to_lds_b128 %0, %1, off"
               :: "v"(lds_off), "v"((unsigned long long)(size_t)gptr)
               : "memory");
}
__device__ __forceinline__ void wait_async0() {
  asm volatile("s_wait_asynccnt 0" ::: "memory");
}

// ------------------------- conversion kernels -------------------------
__global__ void cvt_f32_bf16(const float* __restrict__ in, bf16* __restrict__ out, size_t n) {
  size_t i = (size_t)blockIdx.x * blockDim.x + threadIdx.x;
  if (i < n) out[i] = (bf16)in[i];
}

__global__ void zero_pad(float* __restrict__ p) { p[threadIdx.x] = 0.0f; }

// conv_w [o][i][4] f32  ->  [tap][o][i] bf16
__global__ void conv_repack(const float* __restrict__ cw, bf16* __restrict__ out) {
  size_t idx = (size_t)blockIdx.x * blockDim.x + threadIdx.x;   // < 4*1024*1024
  if (idx >= (size_t)4 * DR * DR) return;
  int tap = (int)(idx >> 20);
  int rem = (int)(idx & ((1u << 20) - 1));
  int o = rem >> 10;
  int i = rem & 1023;
  out[idx] = (bf16)cw[((size_t)o * DR + i) * 4 + tap];
}

// ------------------------- WMMA GEMM  C = A(bf16) * B(bf16,W[N,K])^T -------------------------
// 64x64 tile / block, 128 threads = 4 waves, wave: 16 rows x 64 cols = 4 wmma accs.
// Double-buffered LDS fed by global_load_async_to_lds_b128; one barrier per K-step.
// MODE 0: out_f32 = val + bias                          (final projection)
// MODE 1: n<DR -> out_f32 = gelu(val+b)  else out_bf16  (split h into a_branch / b_branch)
// MODE 2: out_f32 = val; out_bf16 = val                 (bc dual precision, no bias)
// MODE 3: out_f32 = sigmoid(val + b)                    (gates)
// CONV: A indexed with causal tap shift (K = 4*1024); t<0 rows read a zeroed pad buffer.
template <int MODE, bool CONV>
__global__ __launch_bounds__(128) void gemm_wmma(
    const bf16* __restrict__ A, const bf16* __restrict__ B,
    const float* __restrict__ bias,
    float* __restrict__ outF, bf16* __restrict__ outH,
    const bf16* __restrict__ zpad,
    int M, int N, int K, int lda, int ldb)
{
  __shared__ alignas(16) bf16 lA[2][64 * 40];   // stride 40 bf16 = 80B rows -> conflict-free b128
  __shared__ alignas(16) bf16 lB[2][64 * 40];

  const int tid  = threadIdx.x;
  const int wave = tid >> 5;
  const int lane = tid & 31;
  const int ll   = lane & 15;
  const int hi   = lane >> 4;
  const int mbase = blockIdx.y * 64;
  const int nbase = blockIdx.x * 64;

  FragC acc[4];
#pragma unroll
  for (int j = 0; j < 4; ++j)
#pragma unroll
    for (int e = 0; e < 8; ++e) acc[j].f[e] = 0.0f;

  // Issue one 64x32-bf16 tile of A and B into LDS buffer `buf` (4 async instrs / wave).
  auto stage = [&](int k0, int buf) {
    int tap = 0, kk = k0;
    if (CONV) { tap = k0 >> 10; kk = k0 & 1023; }          // 32-wide K-tile never crosses a tap
    const bf16* Bsrc = CONV ? (B + ((size_t)tap << 20)) : B;
#pragma unroll
    for (int it = 0; it < 2; ++it) {
      int u = tid + it * 128;      // 0..255 -> 64 rows x 4 xmm quads
      int r = u >> 2;
      int q = u & 3;
      const bf16* ga;
      int gm = mbase + r + (CONV ? (tap - 3) : 0);
      if (CONV && gm < 0) ga = zpad;                        // causal zero padding
      else                ga = A + (size_t)gm * lda + kk + q * 8;
      async_b128((unsigned)(size_t)&lA[buf][r * 40 + q * 8], ga);
      const bf16* gb = Bsrc + (size_t)(nbase + r) * ldb + kk + q * 8;
      async_b128((unsigned)(size_t)&lB[buf][r * 40 + q * 8], gb);
    }
  };

  int p = 0;
  stage(0, 0);

  for (int k0 = 0; k0 < K; k0 += 32) {
    wait_async0();        // own async writes for buf p landed
    __syncthreads();      // all waves' writes landed; prior reads of buf p^1 retired
    if (k0 + 32 < K) stage(k0 + 32, p ^ 1);   // overlap next tile's DMA with compute

    // A fragment: lane holds row (wave*16 + ll); K = hi*8 + {0..7}, hi*8+16 + {0..7}
    FragAB fa;
    {
      int r = (wave << 4) + ll;
      fa.q[0] = *(const uint4*)(&lA[p][r * 40 + hi * 8]);
      fa.q[1] = *(const uint4*)(&lA[p][r * 40 + hi * 8 + 16]);
    }
#pragma unroll
    for (int j = 0; j < 4; ++j) {
      // B fragment: lane holds col (j*16 + ll); K = hi*8 + {0..7}, hi*8+16 + {0..7}
      FragAB fb;
      int c = (j << 4) + ll;
      fb.q[0] = *(const uint4*)(&lB[p][c * 40 + hi * 8]);
      fb.q[1] = *(const uint4*)(&lB[p][c * 40 + hi * 8 + 16]);
      acc[j].v = __builtin_amdgcn_wmma_f32_16x16x32_bf16(
          false, fa.v, false, fb.v, (short)0, acc[j].v, false, false);
    }
    p ^= 1;
  }

  // epilogue: D layout — lanes 0-15: M=e, lanes 16-31: M=e+8; col = j*16 + ll
#pragma unroll
  for (int j = 0; j < 4; ++j) {
    int n = nbase + (j << 4) + ll;
    float bv = bias ? bias[n] : 0.0f;
#pragma unroll
    for (int e = 0; e < 8; ++e) {
      int m = mbase + (wave << 4) + hi * 8 + e;
      float val = acc[j].f[e] + bv;
      if (MODE == 0) {
        outF[(size_t)m * N + n] = val;
      } else if (MODE == 1) {
        if (n < DR) outF[(size_t)m * DR + n] = gelu_tanh(val);
        else        outH[(size_t)m * DR + (n - DR)] = (bf16)val;
      } else if (MODE == 2) {
        outF[(size_t)m * N + n] = val;
        outH[(size_t)m * N + n] = (bf16)val;
      } else {
        outF[(size_t)m * N + n] = sigm(val);
      }
    }
  }
}

// ------------------------- RG-LRU gates -------------------------
__global__ void gates_kernel(const float* __restrict__ G, const float* __restrict__ bc,
                             const float* __restrict__ Lambda,
                             float* __restrict__ At, float* __restrict__ Gt) {
  size_t i = (size_t)blockIdx.x * blockDim.x + threadIdx.x;   // < T*DR
  if (i >= (size_t)T_LEN * DR) return;
  int t = (int)(i >> 10);
  int c = (int)(i & 1023);
  float inp = G[(size_t)t * (2 * DR) + c];
  float rec = G[(size_t)t * (2 * DR) + DR + c];
  float l  = Lambda[c];
  float sp = (l > 20.0f) ? l : log1pf(expf(l));
  float a  = expf(-8.0f * sp * rec);
  At[i] = a;
  Gt[i] = sqrtf(fmaxf(1.0f - a * a, 0.0f)) * inp * bc[i];
}

// ------------------------- chunked linear scan (3 passes) -------------------------
__global__ void scan_reduce(const float* __restrict__ At, const float* __restrict__ Gt,
                            float* __restrict__ chP, float* __restrict__ chL) {
  int c  = blockIdx.x * blockDim.x + threadIdx.x;   // channel
  int ch = blockIdx.y;                              // chunk
  size_t idx = ((size_t)ch * CHUNK) * DR + c;
  float h = 0.0f, P = 1.0f;
  for (int t = 0; t < CHUNK; ++t) {
    float a = At[idx];
    h = fmaf(a, h, Gt[idx]);
    P *= a;
    idx += DR;
  }
  chP[ch * DR + c] = P;
  chL[ch * DR + c] = h;
}

__global__ void scan_carries(const float* __restrict__ chP, const float* __restrict__ chL,
                             float* __restrict__ carry) {
  int c = blockIdx.x * blockDim.x + threadIdx.x;
  float h = 0.0f;
  for (int ch = 0; ch < NCHUNK; ++ch) {
    carry[ch * DR + c] = h;
    h = chL[ch * DR + c] + chP[ch * DR + c] * h;
  }
}

__global__ void scan_apply(const float* __restrict__ At, const float* __restrict__ Gt,
                           const float* __restrict__ carry,
                           const float* __restrict__ Ab, bf16* __restrict__ Zb) {
  int c  = blockIdx.x * blockDim.x + threadIdx.x;
  int ch = blockIdx.y;
  size_t idx = ((size_t)ch * CHUNK) * DR + c;
  float h = carry[ch * DR + c];
  for (int t = 0; t < CHUNK; ++t) {
    h = fmaf(At[idx], h, Gt[idx]);
    Zb[idx] = (bf16)(Ab[idx] * h);   // fuse a_branch * y -> bf16 input of final GEMM
    idx += DR;
  }
}

// ------------------------- launch -------------------------
extern "C" void kernel_launch(void* const* d_in, const int* in_sizes, int n_in,
                              void* d_out, int out_size, void* d_ws, size_t ws_size,
                              hipStream_t stream) {
  (void)in_sizes; (void)n_in; (void)out_size; (void)ws_size;
  const float* x     = (const float*)d_in[0];
  const float* w1    = (const float*)d_in[1];
  const float* b1    = (const float*)d_in[2];
  const float* convw = (const float*)d_in[3];
  const float* wrg   = (const float*)d_in[4];
  const float* brg   = (const float*)d_in[5];
  const float* wout  = (const float*)d_in[6];
  const float* bout  = (const float*)d_in[7];
  const float* Lam   = (const float*)d_in[8];
  float* out = (float*)d_out;

  char* ws = (char*)d_ws;
  size_t off = 0;
  auto take = [&](size_t bytes) -> char* {
    char* p = ws + off;
    off = (off + bytes + 255) & ~(size_t)255;
    return p;
  };
  bf16*  XB    = (bf16*) take((size_t)T_LEN * DM * 2);
  bf16*  W1B   = (bf16*) take((size_t)2 * DR * DM * 2);
  bf16*  CONVB = (bf16*) take((size_t)4 * DR * DR * 2);
  bf16*  WRGB  = (bf16*) take((size_t)2 * DR * DR * 2);
  bf16*  WOUTB = (bf16*) take((size_t)DM * DR * 2);
  float* AB    = (float*)take((size_t)T_LEN * DR * 4);   // a_branch (f32)
  bf16*  BB    = (bf16*) take((size_t)T_LEN * DR * 2);   // b_branch (bf16)
  float* BCF   = (float*)take((size_t)T_LEN * DR * 4);   // bc f32
  bf16*  BCB   = (bf16*) take((size_t)T_LEN * DR * 2);   // bc bf16
  float* G     = (float*)take((size_t)T_LEN * 2 * DR * 4);
  float* AT    = (float*)take((size_t)T_LEN * DR * 4);
  float* GT    = (float*)take((size_t)T_LEN * DR * 4);
  float* CHP   = (float*)take((size_t)NCHUNK * DR * 4);
  float* CHL   = (float*)take((size_t)NCHUNK * DR * 4);
  float* CARRY = (float*)take((size_t)NCHUNK * DR * 4);
  float* ZPADF = (float*)take(256);                      // zeroed causal pad
  bf16*  ZB    = BCB;   // alias: BCB is dead after the rg-gate GEMM

  auto nb = [](size_t n) { return (unsigned)((n + 255) / 256); };

  zero_pad<<<1, 64, 0, stream>>>(ZPADF);
  cvt_f32_bf16<<<nb((size_t)T_LEN * DM), 256, 0, stream>>>(x,    XB,    (size_t)T_LEN * DM);
  cvt_f32_bf16<<<nb((size_t)2 * DR * DM), 256, 0, stream>>>(w1,  W1B,  (size_t)2 * DR * DM);
  cvt_f32_bf16<<<nb((size_t)2 * DR * DR), 256, 0, stream>>>(wrg, WRGB, (size_t)2 * DR * DR);
  cvt_f32_bf16<<<nb((size_t)DM * DR), 256, 0, stream>>>(wout, WOUTB, (size_t)DM * DR);
  conv_repack <<<nb((size_t)4 * DR * DR), 256, 0, stream>>>(convw, CONVB);

  // 1) h = x @ w1^T + b1 ; split gelu / b_branch
  gemm_wmma<1, false><<<dim3(2 * DR / 64, T_LEN / 64), 128, 0, stream>>>(
      XB, W1B, b1, AB, BB, nullptr, T_LEN, 2 * DR, DM, DM, DM);
  // 2) causal 4-tap channel-mixing conv as K=4096 GEMM
  gemm_wmma<2, true ><<<dim3(DR / 64, T_LEN / 64), 128, 0, stream>>>(
      BB, CONVB, nullptr, BCF, BCB, (const bf16*)ZPADF, T_LEN, DR, 4 * DR, DR, DR);
  // 3) g = sigmoid(bc @ w_rg^T + b_rg)
  gemm_wmma<3, false><<<dim3(2 * DR / 64, T_LEN / 64), 128, 0, stream>>>(
      BCB, WRGB, brg, G, nullptr, nullptr, T_LEN, 2 * DR, DR, DR, DR);
  // 4) gates -> a_t, gated_x ; chunked linear recurrence
  gates_kernel<<<nb((size_t)T_LEN * DR), 256, 0, stream>>>(G, BCF, Lam, AT, GT);
  scan_reduce <<<dim3(DR / 256, NCHUNK), 256, 0, stream>>>(AT, GT, CHP, CHL);
  scan_carries<<<DR / 256, 256, 0, stream>>>(CHP, CHL, CARRY);
  scan_apply  <<<dim3(DR / 256, NCHUNK), 256, 0, stream>>>(AT, GT, CARRY, AB, ZB);
  // 5) out = (a_branch * y) @ w_out^T + b_out
  gemm_wmma<0, false><<<dim3(DM / 64, T_LEN / 64), 128, 0, stream>>>(
      ZB, WOUTB, bout, out, nullptr, nullptr, T_LEN, DM, DR, DR, DR);
}
